// LocContrastiveLoss_72636486910306
// MI455X (gfx1250) — compile-verified, matrix operand order checked by reading, not run on hardware
//
#include <hip/hip_runtime.h>
#include <hip/hip_bf16.h>
#include <math.h>

// ---------------- problem constants ----------------
#define BB 4
#define CC 256
#define HH 360
#define WW 360
#define HWSZ (HH * WW)           // 129600 (multiple of 32 -> waves are batch-uniform)
#define NBOX 50
#define TOPK_GEO 20
#define TOPK_STATIC 15
#define NUM_AMB 30
#define TEMP_INV 10.0f           // 1/0.1
#define MARGIN 0.3f

// padded matrix dims for WMMA
#define QROWS 128                // amb 120 -> 128
#define PROWS 80                 // geo 80 (exact)
#define SROWS 64                 // static 60 -> 64
#define SVALID 60

#define NEG_INF (-__builtin_huge_valf())

typedef __attribute__((ext_vector_type(16))) _Float16 v16h;
typedef __attribute__((ext_vector_type(8)))  float    v8f;

// ---------------- workspace layout (bytes) ----------------
#define WS_INTEN 0                      // BB*HWSZ f32          = 2,073,600
#define WS_CVAL  2073600                // BB*HWSZ f32          = 2,073,600
#define WS_CIDX  4147200                // BB*HWSZ i32          = 2,073,600
#define WS_CNT   6220800                // BB i32 (pad to 32)
#define WS_GIDX  6220832                // BB*TOPK_GEO i32      = 320
#define WS_SIDX  6221152                // BB*TOPK_STATIC i32   = 240 (pad)
#define WS_QMAT  6221408                // 128*256 f16          = 65,536
#define WS_PMAT  6286944                // 80*256 f16           = 40,960
#define WS_SMAT  6327904                // 64*256 f16           = 32,768

// ============== kernel 1: channel-norm intensity (the 531 MB streaming pass) ==============
__global__ void __launch_bounds__(256) intensity_kernel(const float* __restrict__ loc,
                                                        float* __restrict__ inten,
                                                        int* __restrict__ count) {
    // zero the per-batch candidate counters for this call (stream-ordered before compaction)
    if (blockIdx.x == 0 && threadIdx.x < BB) count[threadIdx.x] = 0;

    int t = blockIdx.x * blockDim.x + threadIdx.x;
    if (t >= BB * (HWSZ / 4)) return;
    int b = t / (HWSZ / 4);
    int p = (t % (HWSZ / 4)) * 4;
    const float* base = loc + (size_t)b * CC * HWSZ + p;
    float ax = 0.f, ay = 0.f, az = 0.f, aw = 0.f;
#pragma unroll 8
    for (int c = 0; c < CC; ++c) {
        float4 v = *(const float4*)(base + (size_t)c * HWSZ);
        ax += v.x * v.x; ay += v.y * v.y; az += v.z * v.z; aw += v.w * v.w;
    }
    float4 o = make_float4(sqrtf(ax), sqrtf(ay), sqrtf(az), sqrtf(aw));
    *(float4*)(inten + (size_t)b * HWSZ + p) = o;
}

// ============== kernel 2: 3x3 local-max test + stream-compaction of candidates ==============
__global__ void __launch_bounds__(256) mask_compact_kernel(const float* __restrict__ inten,
                                                           float* __restrict__ cval,
                                                           int* __restrict__ cidx,
                                                           int* __restrict__ count) {
    int t = blockIdx.x * blockDim.x + threadIdx.x;
    if (t >= BB * HWSZ) return;
    int b = t / HWSZ;                 // uniform within a wave (HWSZ % 32 == 0)
    int p = t % HWSZ;
    int y = p / WW, x = p % WW;
    const float* ib = inten + (size_t)b * HWSZ;
    float v = ib[p];
    float pool = NEG_INF;
#pragma unroll
    for (int dy = -1; dy <= 1; ++dy) {
        int yy = y + dy;
        if (yy < 0 || yy >= HH) continue;
#pragma unroll
        for (int dx = -1; dx <= 1; ++dx) {
            int xx = x + dx;
            if (xx < 0 || xx >= WW) continue;
            pool = fmaxf(pool, ib[yy * WW + xx]);
        }
    }
    bool take = (v == pool);

    // wave-aggregated compaction: one atomic per wave32
    unsigned mask32 = (unsigned)__ballot(take);
    int lane = threadIdx.x & 31;
    int nsel = __popc(mask32);
    int base = 0;
    if (nsel > 0) {
        int leader = __ffs(mask32) - 1;
        if (lane == leader) base = atomicAdd(&count[b], nsel);
        base = __shfl(base, leader, 32);
        if (take) {
            int off = __popc(mask32 & ((1u << lane) - 1u));
            size_t slot = (size_t)b * HWSZ + base + off;
            cval[slot] = v;
            cidx[slot] = p;
        }
    }
}

// ============== kernel 3: top-20 over the compacted candidates (one block per batch) ==============
// Deterministic despite nondeterministic compaction order: argmax tie-broken on
// original pixel index (unique per candidate), matching jax top_k's lowest-index rule.
__global__ void __launch_bounds__(1024) topk_kernel(float* __restrict__ cval,
                                                    const int* __restrict__ cidx,
                                                    const int* __restrict__ count,
                                                    int* __restrict__ geo_idx) {
    int b = blockIdx.x;
    int cnt = count[b];
    if (cnt > HWSZ) cnt = HWSZ;
    float* cv = cval + (size_t)b * HWSZ;
    const int* ci = cidx + (size_t)b * HWSZ;
    __shared__ float sv[1024];
    __shared__ int   si[1024];
    __shared__ int   sp[1024];
    for (int it = 0; it < TOPK_GEO; ++it) {
        float bv = NEG_INF; int bi = 0x7fffffff; int bp = -1;
        for (int i = threadIdx.x; i < cnt; i += 1024) {
            float v = cv[i];
            if (v == NEG_INF) continue;        // already selected
            int idx = ci[i];
            if (v > bv || (v == bv && idx < bi)) { bv = v; bi = idx; bp = i; }
        }
        sv[threadIdx.x] = bv; si[threadIdx.x] = bi; sp[threadIdx.x] = bp;
        __syncthreads();
        for (int s = 512; s > 0; s >>= 1) {
            if (threadIdx.x < s) {
                float ov = sv[threadIdx.x + s]; int oi = si[threadIdx.x + s];
                if (ov > sv[threadIdx.x] ||
                    (ov == sv[threadIdx.x] && oi < si[threadIdx.x])) {
                    sv[threadIdx.x] = ov; si[threadIdx.x] = oi;
                    sp[threadIdx.x] = sp[threadIdx.x + s];
                }
            }
            __syncthreads();
        }
        if (threadIdx.x == 0) {
            geo_idx[b * TOPK_GEO + it] = (sp[0] >= 0) ? si[0] : 0;
            if (sp[0] >= 0) cv[sp[0]] = NEG_INF;
        }
        __syncthreads();
    }
}

// ============== kernel 4: static box-center flat indices ==============
__global__ void staticidx_kernel(const float* __restrict__ gt_boxes,
                                 const int* __restrict__ box_perm,
                                 int* __restrict__ sidx) {
    int t = threadIdx.x;
    if (t >= BB * TOPK_STATIC) return;
    int b = t / TOPK_STATIC, k = t % TOPK_STATIC;
    int perm = box_perm[b * TOPK_STATIC + k];
    float x = gt_boxes[((size_t)b * NBOX + perm) * 8 + 0];
    float y = gt_boxes[((size_t)b * NBOX + perm) * 8 + 1];
    float cxf = fminf(fmaxf((x + 59.9f) / 119.8f * (float)WW, 0.f), (float)(WW - 1));
    float cyf = fminf(fmaxf((y + 59.9f) / 119.8f * (float)HH, 0.f), (float)(HH - 1));
    int cx = (int)cxf, cy = (int)cyf;
    sidx[t] = cy * WW + cx;
}

// ============== kernel 5: gather + L2-normalize + f16 pack (one wave32 per row) ==============
__global__ void __launch_bounds__(256) gather_norm_kernel(const float* __restrict__ loc,
                                                          const int* __restrict__ amb,
                                                          const int* __restrict__ geo_idx,
                                                          const int* __restrict__ stat_idx,
                                                          _Float16* __restrict__ Q,
                                                          _Float16* __restrict__ P,
                                                          _Float16* __restrict__ S) {
    int wid  = (blockIdx.x * blockDim.x + threadIdx.x) >> 5;   // global wave id = row id
    int lane = threadIdx.x & 31;
    if (wid >= QROWS + PROWS + SROWS) return;

    _Float16* dst;
    int b = 0, idx = 0;
    bool zero = false;
    if (wid < QROWS) {
        dst = Q + (size_t)wid * CC;
        if (wid < BB * NUM_AMB) { b = wid / NUM_AMB; idx = amb[b * NUM_AMB + wid % NUM_AMB]; }
        else zero = true;
    } else if (wid < QROWS + PROWS) {
        int r = wid - QROWS;
        dst = P + (size_t)r * CC;
        b = r / TOPK_GEO; idx = geo_idx[b * TOPK_GEO + r % TOPK_GEO];
    } else {
        int r = wid - QROWS - PROWS;
        dst = S + (size_t)r * CC;
        if (r < SVALID) { b = r / TOPK_STATIC; idx = stat_idx[b * TOPK_STATIC + r % TOPK_STATIC]; }
        else zero = true;
    }

    float v[8];
    float ssq = 0.f;
    if (!zero) {
        const float* base = loc + (size_t)b * CC * HWSZ + idx;
#pragma unroll
        for (int i = 0; i < 8; ++i) {
            v[i] = base[(size_t)(lane + 32 * i) * HWSZ];
            ssq += v[i] * v[i];
        }
    } else {
#pragma unroll
        for (int i = 0; i < 8; ++i) v[i] = 0.f;
    }
#pragma unroll
    for (int off = 1; off < 32; off <<= 1) ssq += __shfl_xor(ssq, off, 32);
    float scale = zero ? 0.f : 1.f / fmaxf(sqrtf(ssq), 1e-12f);
#pragma unroll
    for (int i = 0; i < 8; ++i) dst[lane + 32 * i] = (_Float16)(v[i] * scale);
}

// ============== kernel 6: WMMA similarity GEMMs + hinge loss ==============
// Fragment layout per CDNA5 ISA 16-bit 16x32 A-matrix table (lanes 0-15 = row,
// K pairs per VGPR, +8 K offset for lanes 16-31). Same row-major pattern loads
// the B operand of Q·Kᵀ from the K-matrix rows.
__device__ inline v16h load_frag(const _Float16* __restrict__ mat, int row0, int k0, int lane) {
    int r = row0 + (lane & 15);
    const _Float16* p = mat + (size_t)r * CC + k0 + ((lane >> 4) << 3);
    v16h a;
#pragma unroll
    for (int i = 0; i < 8; ++i) {
        int kb = (i < 4) ? (2 * i) : (2 * i + 8);
        a[2 * i]     = p[kb];
        a[2 * i + 1] = p[kb + 1];
    }
    return a;
}

__global__ void __launch_bounds__(256) gemm_loss_kernel(const _Float16* __restrict__ Q,
                                                        const _Float16* __restrict__ P,
                                                        const _Float16* __restrict__ S,
                                                        float* __restrict__ out) {
    __shared__ float spos[QROWS];
    __shared__ float sneg[QROWS];
    int tid  = threadIdx.x;
    int wave = tid >> 5;
    int lane = tid & 31;
    int m0   = wave * 16;

    float runpos[8], runneg[8];
#pragma unroll
    for (int j = 0; j < 8; ++j) { runpos[j] = NEG_INF; runneg[j] = NEG_INF; }

    // ---- sim_pos: Q[128x256] x P^T[256x80], running per-(m, n%16) max ----
#pragma unroll
    for (int nt = 0; nt < PROWS / 16; ++nt) {
        v8f c = {};
#pragma unroll
        for (int kc = 0; kc < CC / 32; ++kc) {
            v16h a = load_frag(Q, m0, kc * 32, lane);
            v16h b = load_frag(P, nt * 16, kc * 32, lane);
            c = __builtin_amdgcn_wmma_f32_16x16x32_f16(false, a, false, b, (short)0, c, false, false);
        }
#pragma unroll
        for (int j = 0; j < 8; ++j) runpos[j] = fmaxf(runpos[j], c[j]);
    }

    // ---- sim_neg: Q x S^T (64 cols, 60 valid; mask padded cols in last tile) ----
#pragma unroll
    for (int nt = 0; nt < SROWS / 16; ++nt) {
        v8f c = {};
#pragma unroll
        for (int kc = 0; kc < CC / 32; ++kc) {
            v16h a = load_frag(Q, m0, kc * 32, lane);
            v16h b = load_frag(S, nt * 16, kc * 32, lane);
            c = __builtin_amdgcn_wmma_f32_16x16x32_f16(false, a, false, b, (short)0, c, false, false);
        }
#pragma unroll
        for (int j = 0; j < 8; ++j) {
            float v = c[j];
            if (nt == (SROWS / 16 - 1) && (lane & 15) >= (SVALID - (SROWS - 16))) v = NEG_INF;
            runneg[j] = fmaxf(runneg[j], v);
        }
    }

    // ---- cross-lane row max: C layout lane = N (per half), VGPR j = row j / j+8 ----
#pragma unroll
    for (int j = 0; j < 8; ++j) {
        float vp = runpos[j], vn = runneg[j];
#pragma unroll
        for (int off = 1; off < 16; off <<= 1) {
            vp = fmaxf(vp, __shfl_xor(vp, off, 32));
            vn = fmaxf(vn, __shfl_xor(vn, off, 32));
        }
        if ((lane & 15) == 0) {
            int m = m0 + j + ((lane >> 4) << 3);
            spos[m] = vp;
            sneg[m] = vn;
        }
    }
    __syncthreads();

    if (tid == 0) {
        float acc = 0.f;
        for (int r = 0; r < BB * NUM_AMB; ++r) {
            float l = MARGIN + TEMP_INV * (sneg[r] - spos[r]);
            acc += fmaxf(l, 0.f);
        }
        out[0] = acc / (float)(BB * NUM_AMB);
    }
}

// ============== host-side launcher ==============
extern "C" void kernel_launch(void* const* d_in, const int* in_sizes, int n_in,
                              void* d_out, int out_size, void* d_ws, size_t ws_size,
                              hipStream_t stream) {
    const float* loc      = (const float*)d_in[0];   // [4,256,360,360] f32
    const float* gt_boxes = (const float*)d_in[1];   // [4,50,8] f32
    const int*   amb_idx  = (const int*)d_in[2];     // [4,30] i32
    const int*   box_perm = (const int*)d_in[3];     // [4,15] i32
    float* out = (float*)d_out;

    char* ws = (char*)d_ws;
    float*    inten    = (float*)(ws + WS_INTEN);
    float*    cval     = (float*)(ws + WS_CVAL);
    int*      cidx     = (int*)(ws + WS_CIDX);
    int*      count    = (int*)(ws + WS_CNT);
    int*      geo_idx  = (int*)(ws + WS_GIDX);
    int*      stat_idx = (int*)(ws + WS_SIDX);
    _Float16* Qmat     = (_Float16*)(ws + WS_QMAT);
    _Float16* Pmat     = (_Float16*)(ws + WS_PMAT);
    _Float16* Smat     = (_Float16*)(ws + WS_SMAT);

    // 1) streaming channel-norm (the only bandwidth-relevant pass: ~531 MB) + counter reset
    {
        int threads = BB * (HWSZ / 4);
        intensity_kernel<<<(threads + 255) / 256, 256, 0, stream>>>(loc, inten, count);
    }
    // 2) 3x3 local-max + wave-aggregated stream compaction (~14k candidates/batch)
    {
        int threads = BB * HWSZ;
        mask_compact_kernel<<<(threads + 255) / 256, 256, 0, stream>>>(inten, cval, cidx, count);
    }
    // 3) per-batch top-20 over compacted candidates only
    topk_kernel<<<BB, 1024, 0, stream>>>(cval, cidx, count, geo_idx);
    // 4) static box-center indices
    staticidx_kernel<<<1, 64, 0, stream>>>(gt_boxes, box_perm, stat_idx);
    // 5) gather + normalize + f16 pack (272 rows, one wave32 each)
    gather_norm_kernel<<<(QROWS + PROWS + SROWS) / 8, 256, 0, stream>>>(
        loc, amb_idx, geo_idx, stat_idx, Qmat, Pmat, Smat);
    // 6) WMMA similarity GEMMs + hinge loss (single workgroup, 8 waves)
    gemm_loss_kernel<<<1, 256, 0, stream>>>(Qmat, Pmat, Smat, out);
}
